// S2SModel_79963701117700
// MI455X (gfx1250) — matrix operand order repaired and checked by reading, not compiled, not run on hardware
//
#include <hip/hip_runtime.h>
#include <cstdint>

#define DEVINL __device__ __forceinline__

typedef __attribute__((ext_vector_type(16))) _Float16 v16h;
typedef __attribute__((ext_vector_type(8)))  _Float16 v8h;
typedef __attribute__((ext_vector_type(8)))  float    v8f;

constexpr int BB = 32;    // batch
constexpr int TT = 32;    // seq len
constexpr int HH = 512;   // hidden == embed dim
constexpr int VV = 6000;  // vocab (375 * 16)
constexpr int SS = 6;     // sentences
constexpr int G4H = 2048; // 4*H

DEVINL v8f wmma16(v16h a, v16h b, v8f c) {
  // D = A(16x32 f16) * B(32x16 f16) + C(16x16 f32)
  return __builtin_amdgcn_wmma_f32_16x16x32_f16(false, a, false, b, (short)0, c,
                                                false, false);
}

// A fragment from f16 row-major A (leading dim lda halfs), rows [m0,m0+16),
// K-chunk [kk,kk+32). ISA layout: lanes 0-15 kbase=0, lanes 16-31 kbase=8;
// per lane e=0..7 -> K=kbase+e, e=8..15 -> K=kbase+16+(e-8): two 16B runs.
DEVINL v16h load_a16(const _Float16* __restrict__ A, int lda, int m0, int kk, int lane) {
  const _Float16* r = A + (size_t)(m0 + (lane & 15)) * lda + kk + ((lane & 16) ? 8 : 0);
  v8h lo = *(const v8h*)(r);
  v8h hi = *(const v8h*)(r + 16);
  v16h a;
#pragma unroll
  for (int i = 0; i < 8; ++i) { a[i] = lo[i]; a[8 + i] = hi[i]; }
  return a;
}

// B fragment from f16 weight W stored [N,K] row-major (ldw halfs): same layout
// as A: lanes 0-15 kbase=0, lanes 16-31 kbase=8; e<8 -> kbase+e, else kbase+16+(e-8).
DEVINL v16h load_b(const _Float16* __restrict__ W, int ldw, int n, int kk, int lane) {
  const _Float16* r = W + (size_t)n * ldw + kk + ((lane & 16) ? 8 : 0);
  v8h lo = *(const v8h*)(r);
  v8h hi = *(const v8h*)(r + 16);
  v16h b;
#pragma unroll
  for (int i = 0; i < 8; ++i) { b[i] = lo[i]; b[8 + i] = hi[i]; }
  return b;  // two global_load_b128
}

DEVINL float sigf(float x) { return 1.f / (1.f + expf(-x)); }

// ---------------- elementwise helpers ----------------

__global__ void cvt_k(const float* __restrict__ in, _Float16* __restrict__ o, int n) {
  int i = blockIdx.x * 256 + threadIdx.x;
  if (i < n) o[i] = (_Float16)in[i];
}

__global__ void zero_k(float* __restrict__ p, int n) {
  int i = blockIdx.x * 256 + threadIdx.x;
  if (i < n) p[i] = 0.f;
}

__global__ void zero16_k(_Float16* __restrict__ p, int n) {
  int i = blockIdx.x * 256 + threadIdx.x;
  if (i < n) p[i] = (_Float16)0.f;
}

// out16[(t*BB+b)*HH + h] = (f16) emb[tok[b,t]]
__global__ void embed_enc_k(const float* __restrict__ emb, const int* __restrict__ tok,
                            int strideB, _Float16* __restrict__ outp) {
  int row = blockIdx.x;              // t*BB + b
  int t = row >> 5, b = row & 31;
  int token = tok[b * strideB + t];
  const float* src = emb + (size_t)token * HH;
  _Float16* dst = outp + (size_t)row * HH;
  for (int h = threadIdx.x; h < HH; h += 256) dst[h] = (_Float16)src[h];
}

// decoder input tokens are [SOS, tok[:, :-1]]
__global__ void embed_dec_k(const float* __restrict__ emb, const int* __restrict__ tok,
                            int strideB, _Float16* __restrict__ outp) {
  int row = blockIdx.x;              // t*BB + b
  int t = row >> 5, b = row & 31;
  int token = (t == 0) ? 1 : tok[b * strideB + (t - 1)];
  const float* src = emb + (size_t)token * HH;
  _Float16* dst = outp + (size_t)row * HH;
  for (int h = threadIdx.x; h < HH; h += 256) dst[h] = (_Float16)src[h];
}

// mask[b,t] = any_{j<t} tok[b,j] in {EOS1=2, EOS2=3}
__global__ void mask_k(const int* __restrict__ tok, int strideB, int* __restrict__ mask) {
  int b = blockIdx.x;
  int t = threadIdx.x;
  int seen = 0;
  for (int j = 0; j < t; ++j) {
    int v = tok[b * strideB + j];
    seen |= (v == 2) | (v == 3);
  }
  mask[b * TT + t] = seen;
}

// LayerNorm over H=512; rows (t*BB+b); transpose=1 writes row (b*TT+t).
// Writes f32 result and an f16 mirror (GEMM A operand).
__global__ void layernorm_k(const float* __restrict__ in, const float* __restrict__ gg,
                            const float* __restrict__ bb, float* __restrict__ outp,
                            _Float16* __restrict__ out16, int transpose) {
  __shared__ float red[256];
  int row = blockIdx.x;
  int tid = threadIdx.x;
  const float* r = in + (size_t)row * HH;
  float x0 = r[tid], x1 = r[tid + 256];
  red[tid] = x0 + x1;
  __syncthreads();
  for (int off = 128; off > 0; off >>= 1) {
    if (tid < off) red[tid] += red[tid + off];
    __syncthreads();
  }
  float mean = red[0] * (1.f / HH);
  __syncthreads();
  float d0 = x0 - mean, d1 = x1 - mean;
  red[tid] = d0 * d0 + d1 * d1;
  __syncthreads();
  for (int off = 128; off > 0; off >>= 1) {
    if (tid < off) red[tid] += red[tid + off];
    __syncthreads();
  }
  float rstd = rsqrtf(red[0] * (1.f / HH) + 1e-5f);
  int t = row >> 5, b = row & 31;
  int orow = transpose ? (b * TT + t) : row;
  float y0 = d0 * rstd * gg[tid] + bb[tid];
  float y1 = d1 * rstd * gg[tid + 256] + bb[tid + 256];
  outp[(size_t)orow * HH + tid]        = y0;
  outp[(size_t)orow * HH + tid + 256]  = y1;
  out16[(size_t)orow * HH + tid]       = (_Float16)y0;
  out16[(size_t)orow * HH + tid + 256] = (_Float16)y1;
}

// ---------------- WMMA GEMM kernels ----------------

// C[M,N](f32) = A[M,K](f16) * W[N,K](f16)^T (+ bias[n]); one 16x16 tile per wave.
__global__ void gemm_wmma_bias_k(const _Float16* __restrict__ A, int lda,
                                 const _Float16* __restrict__ W, int ldw,
                                 const float* __restrict__ bias,
                                 float* __restrict__ C, int ldc, int K) {
  int lane = threadIdx.x;
  int n0 = blockIdx.x * 16, m0 = blockIdx.y * 16;
  int n = n0 + (lane & 15);
  v8f acc = {};
  for (int kk = 0; kk < K; kk += 32) {
    v16h bf = load_b(W, ldw, n, kk, lane);
    v16h af = load_a16(A, lda, m0, kk, lane);
    acc = wmma16(af, bf, acc);
  }
  float bv = bias ? bias[n] : 0.f;
  int mb = m0 + ((lane & 16) ? 8 : 0);
#pragma unroll
  for (int rI = 0; rI < 8; ++rI) C[(size_t)(mb + rI) * ldc + n] = acc[rI] + bv;
}

// logits GEMM: A = dec_ln16 [1024,512] rows (t*BB+b); 4 waves/block, each wave a
// distinct n-tile. The shared 16x512 A tile (contiguous 16 KB) is staged into
// LDS with CDNA5 async global->LDS DMA (ASYNCcnt), then read via ds_load.
// Writes out[b, sm1, t, n] directly.
__global__ void __launch_bounds__(128) gemm_wmma_fc_k(
    const _Float16* __restrict__ A,
    const _Float16* __restrict__ W,
    const float* __restrict__ bias,
    float* __restrict__ outp, int sm1) {
  __shared__ __align__(16) _Float16 ldsA[16 * HH];  // 16 KB A tile
  int tid = threadIdx.x;
  int lane = tid & 31, wv = tid >> 5;
  int m0 = blockIdx.y * 16;

  {
    const char* gsrc = (const char*)(A + (size_t)m0 * HH);
    // low 32 bits of a flat LDS pointer are the LDS byte address (aperture rule)
    uint32_t lbase = (uint32_t)(uintptr_t)(void*)&ldsA[0];
#pragma unroll
    for (int c = 0; c < 8; ++c) {                    // 128 thr * 8 * 16 B = 16 KB
      int byteoff = (tid + c * 128) * 16;
      uint32_t ldst = lbase + (uint32_t)byteoff;
      uint64_t gaddr = (uint64_t)(uintptr_t)(gsrc + byteoff);
      asm volatile("global_load_async_to_lds_b128 %0, %1, off"
                   :: "v"(ldst), "v"(gaddr) : "memory");
    }
    asm volatile("s_wait_asynccnt 0x0" ::: "memory");
  }
  __syncthreads();

  int n0 = (blockIdx.x * 4 + wv) * 16;
  if (n0 < VV) {
    int n = n0 + (lane & 15);
    const _Float16* wrow = W + (size_t)n * HH;
    const _Float16* arow = &ldsA[0] + (size_t)(lane & 15) * HH + ((lane & 16) ? 8 : 0);
    v8f acc = {};
    for (int kk = 0; kk < HH; kk += 32) {
      __builtin_prefetch(wrow + kk + 256, 0, 1);     // global_prefetch_b8
      v16h bf = load_b(W, HH, n, kk, lane);
      v8h lo = *(const v8h*)(arow + kk);             // ds_load_b128
      v8h hi = *(const v8h*)(arow + kk + 16);
      v16h af;
#pragma unroll
      for (int i = 0; i < 8; ++i) { af[i] = lo[i]; af[8 + i] = hi[i]; }
      acc = wmma16(af, bf, acc);
    }
    float bv = bias[n];
    int mb = m0 + ((lane & 16) ? 8 : 0);
#pragma unroll
    for (int rI = 0; rI < 8; ++rI) {
      int m = mb + rI;
      int t = m >> 5, b = m & 31;
      outp[(((size_t)b * (SS - 1) + sm1) * TT + t) * VV + n] = acc[rI] + bv;
    }
  }
}

// Fused LSTM step (B=32). 4 waves/block: wave g computes gate g (i,f,g,o) for
// 16 hidden columns. Segmented K loop: X0 (K0 cols of Wih), optional X1
// (K1 cols, attention context), then Hprev (Whh). All activations f16, cell
// state f32. Gates exchanged via LDS; pointwise update emits f16 h + f32 c.
__global__ void __launch_bounds__(128) lstm_step_k(
    const _Float16* __restrict__ X0, int lda0, int K0,
    const _Float16* __restrict__ X1, int K1,
    const _Float16* __restrict__ Hprev,
    const _Float16* __restrict__ Wih, int ldw,
    const _Float16* __restrict__ Whh,
    const float* __restrict__ bias,
    const float* __restrict__ Cprev,
    _Float16* __restrict__ Hout, float* __restrict__ Cout,
    float* __restrict__ copy_out) {
  __shared__ float gsh[4][32][16];
  int lane = threadIdx.x & 31;
  int wv = threadIdx.x >> 5;          // gate index
  int h0 = blockIdx.x * 16;
  int n = wv * HH + h0 + (lane & 15); // row of Wih/Whh in [0,4H)
  v8f c0 = {}, c1 = {};
  for (int kk = 0; kk < K0; kk += 32) {
    v16h bf = load_b(Wih, ldw, n, kk, lane);
    v16h a0 = load_a16(X0, lda0, 0, kk, lane);
    v16h a1 = load_a16(X0, lda0, 16, kk, lane);
    c0 = wmma16(a0, bf, c0);
    c1 = wmma16(a1, bf, c1);
  }
  if (K1 > 0) {
    for (int kk = 0; kk < K1; kk += 32) {
      v16h bf = load_b(Wih, ldw, n, K0 + kk, lane);
      v16h a0 = load_a16(X1, HH, 0, kk, lane);
      v16h a1 = load_a16(X1, HH, 16, kk, lane);
      c0 = wmma16(a0, bf, c0);
      c1 = wmma16(a1, bf, c1);
    }
  }
  for (int kk = 0; kk < HH; kk += 32) {
    v16h bf = load_b(Whh, HH, n, kk, lane);
    v16h a0 = load_a16(Hprev, HH, 0, kk, lane);
    v16h a1 = load_a16(Hprev, HH, 16, kk, lane);
    c0 = wmma16(a0, bf, c0);
    c1 = wmma16(a1, bf, c1);
  }
  float bv = bias[n];
  int mb = (lane & 16) ? 8 : 0;
  int col = lane & 15;
#pragma unroll
  for (int rI = 0; rI < 8; ++rI) {
    gsh[wv][mb + rI][col]      = c0[rI] + bv;
    gsh[wv][16 + mb + rI][col] = c1[rI] + bv;
  }
  __syncthreads();
  for (int e = threadIdx.x; e < BB * 16; e += 128) {
    int m = e >> 4, cc = e & 15;
    float iv = gsh[0][m][cc];
    float fv = gsh[1][m][cc];
    float gv = gsh[2][m][cc];
    float ov = gsh[3][m][cc];
    int idx = m * HH + h0 + cc;
    float cn = sigf(fv) * Cprev[idx] + sigf(iv) * tanhf(gv);
    float hn = sigf(ov) * tanhf(cn);
    Cout[idx] = cn;
    Hout[idx] = (_Float16)hn;
    if (copy_out) copy_out[idx] = hn;
  }
}

// ---------------- attention ----------------

// scores[b,t] = mask ? -1e9 : sum_h tanh(qW[b,h] + encW[b,t,h]) * att_v[h]
__global__ void attn_scores_k(const float* __restrict__ qW, const float* __restrict__ encW,
                              const float* __restrict__ av, const int* __restrict__ mask,
                              float* __restrict__ scores) {
  __shared__ float red[128];
  int bt = blockIdx.x;               // b*TT + t
  int b = bt >> 5;
  int tid = threadIdx.x;
  const float* q = qW + (size_t)b * HH;
  const float* e = encW + (size_t)bt * HH;
  float p = 0.f;
  for (int h = tid; h < HH; h += 128) p += tanhf(q[h] + e[h]) * av[h];
  red[tid] = p;
  __syncthreads();
  for (int off = 64; off > 0; off >>= 1) {
    if (tid < off) red[tid] += red[tid + off];
    __syncthreads();
  }
  if (tid == 0) scores[bt] = mask[bt] ? -1e9f : red[0];
}

// softmax over T, then context16[b,h] = (f16) sum_t a[b,t] * enc_outs[b,t,h]
__global__ void attn_context_k(const float* __restrict__ scores,
                               const float* __restrict__ enc_outs,
                               _Float16* __restrict__ ctx16) {
  __shared__ float pr[TT];
  int b = blockIdx.x;
  int tid = threadIdx.x;
  if (tid < TT) pr[tid] = scores[b * TT + tid];
  __syncthreads();
  if (tid == 0) {
    float mx = pr[0];
    for (int j = 1; j < TT; ++j) mx = fmaxf(mx, pr[j]);
    float sm = 0.f;
    for (int j = 0; j < TT; ++j) { pr[j] = expf(pr[j] - mx); sm += pr[j]; }
    float inv = 1.f / sm;
    for (int j = 0; j < TT; ++j) pr[j] *= inv;
  }
  __syncthreads();
  for (int h = tid; h < HH; h += 256) {
    float acc = 0.f;
#pragma unroll
    for (int t = 0; t < TT; ++t) acc += pr[t] * enc_outs[((size_t)b * TT + t) * HH + h];
    ctx16[b * HH + h] = (_Float16)acc;
  }
}

// ---------------- host orchestration ----------------

extern "C" void kernel_launch(void* const* d_in, const int* in_sizes, int n_in,
                              void* d_out, int out_size, void* d_ws, size_t ws_size,
                              hipStream_t stream) {
  (void)in_sizes; (void)n_in; (void)out_size; (void)ws_size;
  const int*   inputs   = (const int*)d_in[0];
  const int*   targets  = (const int*)d_in[1];
  const float* enc_emb  = (const float*)d_in[2];
  const float* enc_Wih  = (const float*)d_in[3];
  const float* enc_Whh  = (const float*)d_in[4];
  const float* enc_bp   = (const float*)d_in[5];
  const float* enc_ln_g = (const float*)d_in[6];
  const float* enc_ln_b = (const float*)d_in[7];
  const float* dec_emb  = (const float*)d_in[8];
  const float* att_W    = (const float*)d_in[9];
  const float* att_b    = (const float*)d_in[10];
  const float* att_v    = (const float*)d_in[11];
  const float* dec_Wih0 = (const float*)d_in[12];
  const float* dec_Wih1 = (const float*)d_in[13];
  const float* dec_Whh  = (const float*)d_in[14];
  const float* dec_bp   = (const float*)d_in[15];
  const float* dec_ln_g = (const float*)d_in[16];
  const float* dec_ln_b = (const float*)d_in[17];
  const float* fc_W     = (const float*)d_in[18];
  const float* fc_b     = (const float*)d_in[19];
  float* out = (float*)d_out;

  char* wp = (char*)d_ws;
  auto alloc = [&](size_t bytes) -> void* {
    void* p = (void*)wp;
    wp += (bytes + 255) & ~(size_t)255;
    return p;
  };

  // f16 weight mirrors (~26 MB -> L2-resident on MI455X's 192 MB L2)
  _Float16* encWih16   = (_Float16*)alloc((size_t)2 * G4H * HH * 2);
  _Float16* encWhh16   = (_Float16*)alloc((size_t)2 * G4H * HH * 2);
  _Float16* decWih0_16 = (_Float16*)alloc((size_t)G4H * 1024 * 2);
  _Float16* decWih1_16 = (_Float16*)alloc((size_t)G4H * HH * 2);
  _Float16* decWhh16   = (_Float16*)alloc((size_t)2 * G4H * HH * 2);
  _Float16* attW16     = (_Float16*)alloc((size_t)HH * 1024 * 2);
  _Float16* fcW16      = (_Float16*)alloc((size_t)VV * HH * 2);
  // f16 activations (GEMM A operands)
  _Float16* eemb16     = (_Float16*)alloc((size_t)TT * BB * HH * 2);
  _Float16* demb16     = (_Float16*)alloc((size_t)TT * BB * HH * 2);
  _Float16* enc_outs16 = (_Float16*)alloc((size_t)BB * TT * HH * 2);
  _Float16* dec_ln16   = (_Float16*)alloc((size_t)TT * BB * HH * 2);
  // f32 activations
  float* enc_raw  = (float*)alloc((size_t)TT * BB * HH * 4);  // [T,B,H] pre-LN
  float* enc_outs = (float*)alloc((size_t)BB * TT * HH * 4);  // [B,T,H] post-LN
  float* encWb    = (float*)alloc((size_t)BB * TT * HH * 4);  // enc_outs@att_We.T + att_b
  float* dec_raw  = (float*)alloc((size_t)TT * BB * HH * 4);
  float* dec_ln32 = (float*)alloc((size_t)TT * BB * HH * 4);
  float* qW       = (float*)alloc((size_t)BB * HH * 4);
  float* scores   = (float*)alloc((size_t)BB * TT * 4);
  int*   maskb    = (int*)alloc((size_t)BB * TT * 4);
  // cell states f32 (8 ping-pong buffers), h states f16 (8) + context16
  float* cst = (float*)alloc((size_t)8 * BB * HH * 4);
  _Float16* hst = (_Float16*)alloc((size_t)9 * BB * HH * 2);
  float* dc0[2] = { cst + 0 * BB * HH, cst + 1 * BB * HH };
  float* dc1[2] = { cst + 2 * BB * HH, cst + 3 * BB * HH };
  float* ec0[2] = { cst + 4 * BB * HH, cst + 5 * BB * HH };
  float* ec1[2] = { cst + 6 * BB * HH, cst + 7 * BB * HH };
  _Float16* dh0[2] = { hst + 0 * BB * HH, hst + 1 * BB * HH };
  _Float16* dh1[2] = { hst + 2 * BB * HH, hst + 3 * BB * HH };
  _Float16* eh0[2] = { hst + 4 * BB * HH, hst + 5 * BB * HH };
  _Float16* eh1[2] = { hst + 6 * BB * HH, hst + 7 * BB * HH };
  _Float16* context16 = hst + 8 * BB * HH;

  auto cvt = [&](const float* s, _Float16* dp, int n) {
    cvt_k<<<dim3((n + 255) / 256), dim3(256), 0, stream>>>(s, dp, n);
  };
  cvt(enc_Wih, encWih16, 2 * G4H * HH);
  cvt(enc_Whh, encWhh16, 2 * G4H * HH);
  cvt(dec_Wih0, decWih0_16, G4H * 1024);
  cvt(dec_Wih1, decWih1_16, G4H * HH);
  cvt(dec_Whh, decWhh16, 2 * G4H * HH);
  cvt(att_W, attW16, HH * 1024);
  cvt(fc_W, fcW16, VV * HH);
  zero_k<<<dim3((8 * BB * HH + 255) / 256), dim3(256), 0, stream>>>(cst, 8 * BB * HH);
  zero16_k<<<dim3((9 * BB * HH + 255) / 256), dim3(256), 0, stream>>>(hst, 9 * BB * HH);

  int pd0 = 0, pd1 = 0, pe0 = 0, pe1 = 0;
  for (int s = 0; s < SS; ++s) {
    const int* tok;
    int strideB;
    if (s == 0) { tok = inputs; strideB = SS * TT; }
    else        { tok = targets + (size_t)(s - 1) * TT; strideB = (SS - 1) * TT; }

    // ---- decode sentence s ----
    embed_dec_k<<<dim3(TT * BB), dim3(256), 0, stream>>>(dec_emb, tok, strideB, demb16);
    for (int t = 0; t < TT; ++t) {
      if (s > 0) {
        // qW = h1 @ att_W[:, :H].T   (enc part hoisted into encWb per sentence)
        gemm_wmma_bias_k<<<dim3(HH / 16, BB / 16), dim3(32), 0, stream>>>(
            dh1[pd1], HH, attW16, 1024, nullptr, qW, HH, HH);
        attn_scores_k<<<dim3(BB * TT), dim3(128), 0, stream>>>(qW, encWb, att_v, maskb, scores);
        attn_context_k<<<dim3(BB), dim3(256), 0, stream>>>(scores, enc_outs, context16);
      } // s==0: context16 stays all-zero (matches reference)
      lstm_step_k<<<dim3(HH / 16), dim3(128), 0, stream>>>(
          demb16 + (size_t)t * BB * HH, HH, HH, context16, HH,
          dh0[pd0], decWih0_16, 1024, decWhh16, dec_bp, dc0[pd0],
          dh0[pd0 ^ 1], dc0[pd0 ^ 1], nullptr);
      pd0 ^= 1;
      lstm_step_k<<<dim3(HH / 16), dim3(128), 0, stream>>>(
          dh0[pd0], HH, HH, nullptr, 0,
          dh1[pd1], decWih1_16, HH, decWhh16 + (size_t)G4H * HH, dec_bp + G4H, dc1[pd1],
          dh1[pd1 ^ 1], dc1[pd1 ^ 1], dec_raw + (size_t)t * BB * HH);
      pd1 ^= 1;
    }
    if (s > 0) {  // reference discards s==0 logits -> skip LN+fc
      layernorm_k<<<dim3(TT * BB), dim3(256), 0, stream>>>(
          dec_raw, dec_ln_g, dec_ln_b, dec_ln32, dec_ln16, 0);
      gemm_wmma_fc_k<<<dim3((VV / 16 + 3) / 4, (TT * BB) / 16), dim3(128), 0, stream>>>(
          dec_ln16, fcW16, fc_b, out, s - 1);
    }

    // ---- mask + encoder for the next sentence ----
    if (s < SS - 1) {
      mask_k<<<dim3(BB), dim3(TT), 0, stream>>>(tok, strideB, maskb);
      embed_enc_k<<<dim3(TT * BB), dim3(256), 0, stream>>>(enc_emb, tok, strideB, eemb16);
      for (int t = 0; t < TT; ++t) {
        lstm_step_k<<<dim3(HH / 16), dim3(128), 0, stream>>>(
            eemb16 + (size_t)t * BB * HH, HH, HH, nullptr, 0,
            eh0[pe0], encWih16, HH, encWhh16, enc_bp, ec0[pe0],
            eh0[pe0 ^ 1], ec0[pe0 ^ 1], nullptr);
        pe0 ^= 1;
        lstm_step_k<<<dim3(HH / 16), dim3(128), 0, stream>>>(
            eh0[pe0], HH, HH, nullptr, 0,
            eh1[pe1], encWih16 + (size_t)G4H * HH, HH, encWhh16 + (size_t)G4H * HH,
            enc_bp + G4H, ec1[pe1],
            eh1[pe1 ^ 1], ec1[pe1 ^ 1], enc_raw + (size_t)t * BB * HH);
        pe1 ^= 1;
      }
      layernorm_k<<<dim3(TT * BB), dim3(256), 0, stream>>>(
          enc_raw, enc_ln_g, enc_ln_b, enc_outs, enc_outs16, 1);
      // hoisted attention term: encWb = enc_outs @ att_W[:, H:].T + att_b
      gemm_wmma_bias_k<<<dim3(HH / 16, (BB * TT) / 16), dim3(32), 0, stream>>>(
          enc_outs16, HH, attW16 + HH, 1024, att_b, encWb, HH, HH);
    }
  }
}
